// HEGNN_Layer_27384711479753
// MI455X (gfx1250) — matrix operand order repaired and checked by reading, not compiled, not run on hardware
//
#include <hip/hip_runtime.h>
#include <hip/hip_bf16.h>

// ---------------------------------------------------------------------------
// HEGNN layer for MI455X (gfx1250): all matmuls via v_wmma_f32_16x16x32_f16.
// B matrices are pre-swizzled to fragment-major so each lane loads its whole
// 16-half B fragment as one aligned 32-byte vector load.
// ---------------------------------------------------------------------------

typedef __attribute__((ext_vector_type(16))) _Float16 v16h;
typedef __attribute__((ext_vector_type(8)))  float    v8f;

#define NNODE 20000
#define NEDGE 640000
#define HDIM  128
#define RDIM  16
#define SHD   9
#define K1P   288          // 275 padded up to a multiple of 32

// f16 weight pack offsets (in halves), stored in d_ws after the counts array.
// All big matrices are swizzled: offset = ((kc*8 + nt)*32 + lane)*16 + e,
// element = W[kc*32 + (lane>>4)*16 + e][nt*16 + (lane&15)].
#define OFF_W1    0        // [288][128]  (rows >=275 zero)
#define OFF_W2    36864    // [128][128]
#define OFF_WP1   53248    // [128][128]
#define OFF_WV1   69632    // [128][128]
#define OFF_WS1   86016    // [128][128]
#define OFF_WSM   102400   // 3 x [128][16] (pos2 / vel2 / sh2, zero-padded cols)
#define OFF_WNF1  108544   // [256][128]
#define OFF_WNF2  141312   // [128][128]
#define W_TOTAL   157696

#define WS_COUNTS_BYTES 80000   // NNODE * 4, 64B aligned

// ---------------------------------------------------------------------------
// WMMA fragment helpers (wave32).
// ---------------------------------------------------------------------------

// K offset inside a 32-wide chunk for element e of an A fragment, given the
// lane half (0: lanes 0-15, 1: lanes 16-31). k = 16*kh + e: one contiguous
// 16-half run per lane (matches the swizzled B pack).
__device__ __forceinline__ int frag_k(int e, int kh) {
  return (kh << 4) + e;
}

// A tile, row-major [M][lda] f16 (LDS). Row = m0 + (lane&15) for both halves.
__device__ __forceinline__ v16h load_frag_a(const _Float16* A, int lda,
                                            int m0, int k0, int lane) {
  const int kh = lane >> 4;
  const _Float16* p = A + (m0 + (lane & 15)) * lda + k0;
  v16h f;
#pragma unroll
  for (int e = 0; e < 16; ++e) f[e] = p[frag_k(e, kh)];
  return f;
}

// Swizzled B fragment: one aligned 32-byte vector load per lane.
__device__ __forceinline__ v16h load_frag_b_sw(const _Float16* B, int blk,
                                               int lane) {
  return *(const v16h*)(B + (blk * 32 + lane) * 16);
}

__device__ __forceinline__ v8f wmma16(v16h a, v16h b, v8f c) {
  // (neg_a, A, neg_b, B, c_mod, C, reuse_a, reuse_b)
  return __builtin_amdgcn_wmma_f32_16x16x32_f16(false, a, false, b,
                                                (short)0, c, false, false);
}

// Fast silu: v_exp_f32 (TRANS, co-executes with WMMA) + v_rcp_f32 + v_mul.
// Avoids the ~10-op IEEE-precise divide expansion.
__device__ __forceinline__ float silu_f(float x) {
  return x * __builtin_amdgcn_rcpf(1.0f + __expf(-x));
}

// 64xH GEMM shared by all 128-wide layers: wave owns n-tile nt (16 cols),
// 4 M-tiles of 16 rows, accumulating over kchunks*32 of K.
__device__ __forceinline__ void gemm64(const _Float16* A, int lda,
                                       const _Float16* B, int kchunks,
                                       int nt, int lane, v8f acc[4]) {
#pragma unroll
  for (int kc = 0; kc < kchunks; ++kc) {
    v16h bf = load_frag_b_sw(B, kc * 8 + nt, lane);
#pragma unroll
    for (int m = 0; m < 4; ++m) {
      v16h af = load_frag_a(A, lda, m * 16, kc * 32, lane);
      acc[m] = wmma16(af, bf, acc[m]);
    }
  }
}

__device__ __forceinline__ void store_silu_f16(_Float16* dst, v8f acc[4],
                                               float bias, int colg, int rb) {
#pragma unroll
  for (int m = 0; m < 4; ++m)
#pragma unroll
    for (int i = 0; i < 8; ++i)
      dst[(m * 16 + rb + i) * HDIM + colg] = (_Float16)silu_f(acc[m][i] + bias);
}

// ---------------------------------------------------------------------------
// Kernel 0: zero the output accumulators + counts (they double as scatter
// buffers; harness does not re-poison between replays so we must re-init).
// ---------------------------------------------------------------------------
__global__ void hegnn_zero_kernel(float* __restrict__ out,
                                  float* __restrict__ counts) {
  int idx = blockIdx.x * blockDim.x + threadIdx.x;
  if (idx < NNODE * 143) out[idx] = 0.0f;
  if (idx < NNODE) counts[idx] = 0.0f;
}

// ---------------------------------------------------------------------------
// Kernel 1: convert/pack all MLP weights to f16, swizzled fragment-major.
// ---------------------------------------------------------------------------
__device__ __forceinline__ float fetch_sw(const float* __restrict__ W,
                                          int idxl, int Krows) {
  int e    = idxl & 15;
  int lane = (idxl >> 4) & 31;
  int blk  = idxl >> 9;
  int nt   = blk & 7;
  int kc   = blk >> 3;
  int r = kc * 32 + ((lane >> 4) << 4) + e;
  int c = nt * 16 + (lane & 15);
  return (r < Krows) ? W[r * 128 + c] : 0.0f;
}

__global__ void hegnn_prep_kernel(const float* __restrict__ W_msg1,
                                  const float* __restrict__ W_msg2,
                                  const float* __restrict__ W_pos1,
                                  const float* __restrict__ W_vel1,
                                  const float* __restrict__ W_sh1,
                                  const float* __restrict__ W_pos2,
                                  const float* __restrict__ W_vel2,
                                  const float* __restrict__ W_sh2,
                                  const float* __restrict__ W_nf1,
                                  const float* __restrict__ W_nf2,
                                  _Float16* __restrict__ hW) {
  int idx = blockIdx.x * blockDim.x + threadIdx.x;
  if (idx >= W_TOTAL) return;
  float v;
  if (idx < OFF_W2)        v = fetch_sw(W_msg1, idx - OFF_W1, 275);
  else if (idx < OFF_WP1)  v = fetch_sw(W_msg2, idx - OFF_W2, 128);
  else if (idx < OFF_WV1)  v = fetch_sw(W_pos1, idx - OFF_WP1, 128);
  else if (idx < OFF_WS1)  v = fetch_sw(W_vel1, idx - OFF_WV1, 128);
  else if (idx < OFF_WSM)  v = fetch_sw(W_sh1, idx - OFF_WS1, 128);
  else if (idx < OFF_WNF1) {
    // small matrices: single n-tile, blk = kc
    int j  = idx - OFF_WSM;
    int h  = j / 2048, jl = j - h * 2048;
    int e  = jl & 15;
    int lane = (jl >> 4) & 31;
    int kc = jl >> 9;
    int r = kc * 32 + ((lane >> 4) << 4) + e;
    int c = lane & 15;
    const float* Ws = (h == 0) ? W_pos2 : ((h == 1) ? W_vel2 : W_sh2);
    int nc = (h == 2) ? 3 : 2;
    v = (c < nc) ? Ws[r * nc + c] : 0.0f;
  }
  else if (idx < OFF_WNF2) v = fetch_sw(W_nf1, idx - OFF_WNF1, 256);
  else                     v = fetch_sw(W_nf2, idx - OFF_WNF2, 128);
  hW[idx] = (_Float16)v;
}

// ---------------------------------------------------------------------------
// Kernel 2: fused per-edge pipeline. One block = 64 edges, 8 waves.
//   gather -> msg MLP (WMMA) -> 3 heads (WMMA) -> gating -> atomic scatter.
// ---------------------------------------------------------------------------
__global__ void __launch_bounds__(256)
hegnn_edge_kernel(const float* __restrict__ node_feat,
                  const float* __restrict__ node_sh,
                  const float* __restrict__ diff_pos,
                  const float* __restrict__ diff_vel,
                  const float* __restrict__ radial,
                  const int*   __restrict__ edge_index,
                  const float* __restrict__ b_msg1,
                  const float* __restrict__ b_msg2,
                  const float* __restrict__ b_pos1,
                  const float* __restrict__ b_pos2,
                  const float* __restrict__ b_vel1,
                  const float* __restrict__ b_vel2,
                  const float* __restrict__ b_sh1,
                  const float* __restrict__ b_sh2,
                  const _Float16* __restrict__ hW,
                  float* __restrict__ msg_sum,   // [N][128] (d_out region 0)
                  float* __restrict__ sh_sum,    // [N][9]
                  float* __restrict__ pos_sum,   // [N][3]
                  float* __restrict__ vel_sum,   // [N][3]
                  float* __restrict__ counts) {  // [N]
  // LDS layout (60,416 B): msg tile overlays the layer-1 A tile.
  __shared__ __align__(16) char smem[60416];
  _Float16* sA  = (_Float16*)smem;              // [64][288] (phases 1-2)
  _Float16* sM  = (_Float16*)smem;              // [64][128] (phases 3+)
  _Float16* sX  = (_Float16*)(smem + 36864);    // [64][128]
  float*    sG  = (float*)(smem + 53248);       // [64][8] gate outputs
  float*    sShd= (float*)(smem + 55296);       // [64][9] sh_r - sh_c
  float*    sDp = (float*)(smem + 57600);       // [64][3]
  float*    sDv = (float*)(smem + 58368);       // [64][3]
  float*    sIp = (float*)(smem + 59136);       // [64][3]
  int*      sRow= (int*)(smem + 59904);         // [64]
  int*      sCol= (int*)(smem + 60160);         // [64]

  const int t    = threadIdx.x;
  const int lane = t & 31;
  const int wave = t >> 5;
  const int kh   = lane >> 4;
  const int e0   = blockIdx.x * 64;

  // --- phase 1a: per-edge scalar gather (64 lanes) ---
  if (t < 64) {
    int e = e0 + t;
    int r = edge_index[e];
    int c = edge_index[NEDGE + e];
    sRow[t] = r;
    sCol[t] = c;
    float ip0 = 0.f, ip1 = 0.f, ip2 = 0.f;
#pragma unroll
    for (int j = 0; j < SHD; ++j) {
      float a = node_sh[r * SHD + j];
      float b = node_sh[c * SHD + j];
      sShd[t * SHD + j] = a - b;
      float p = a * b;
      if (j == 0)      ip0 += p;
      else if (j < 4)  ip1 += p;
      else             ip2 += p;
    }
    sIp[t * 3 + 0] = ip0;
    sIp[t * 3 + 1] = ip1;
    sIp[t * 3 + 2] = ip2;
#pragma unroll
    for (int d = 0; d < 3; ++d) {
      sDp[t * 3 + d] = diff_pos[e * 3 + d];
      sDv[t * 3 + d] = diff_vel[e * 3 + d];
    }
    atomicAdd(&counts[r], 1.0f);
  }
  __syncthreads();

  // --- phase 1b: build msg_in A tile [64][288] f16 cooperatively ---
  for (int idx = t; idx < 64 * K1P; idx += 256) {
    int j = idx / K1P;
    int k = idx - j * K1P;
    float v;
    if (k < 128)       v = node_feat[sRow[j] * HDIM + k];
    else if (k < 256)  v = node_feat[sCol[j] * HDIM + (k - 128)];
    else if (k < 272)  v = radial[(e0 + j) * RDIM + (k - 256)];
    else if (k < 275)  v = sIp[j * 3 + (k - 272)];
    else               v = 0.0f;
    sA[idx] = (_Float16)v;
  }
  __syncthreads();

  const int colg = wave * 16 + (lane & 15);
  const int rb   = kh * 8;
  v8f zero = {};

  // --- phase 2: msg layer 1  [64x288] x [288x128], silu ---
  {
    v8f acc[4] = {zero, zero, zero, zero};
    gemm64(sA, K1P, hW + OFF_W1, 9, wave, lane, acc);
    store_silu_f16(sX, acc, b_msg1[colg], colg, rb);
  }
  __syncthreads();

  // --- phase 3: msg layer 2  [64x128] x [128x128], silu; scatter msg ---
  {
    v8f acc[4] = {zero, zero, zero, zero};
    gemm64(sX, HDIM, hW + OFF_W2, 4, wave, lane, acc);
    float bias = b_msg2[colg];
#pragma unroll
    for (int m = 0; m < 4; ++m)
#pragma unroll
      for (int i = 0; i < 8; ++i) {
        int rloc = m * 16 + rb + i;
        float mv = silu_f(acc[m][i] + bias);
        sM[rloc * HDIM + colg] = (_Float16)mv;           // overlays dead sA
        atomicAdd(&msg_sum[sRow[rloc] * HDIM + colg], mv);
      }
  }
  __syncthreads();

  // --- phase 4: pos / vel / sh heads ---
  const _Float16* w1s[3] = {hW + OFF_WP1, hW + OFF_WV1, hW + OFF_WS1};
  const float*    b1s[3] = {b_pos1, b_vel1, b_sh1};
  const float*    b2s[3] = {b_pos2, b_vel2, b_sh2};
  const int goff[3] = {0, 2, 4};
  const int ncol[3] = {2, 2, 3};
  for (int h = 0; h < 3; ++h) {
    {   // hidden layer: [64x128] x [128x128], silu -> sX
      v8f acc[4] = {zero, zero, zero, zero};
      gemm64(sM, HDIM, w1s[h], 4, wave, lane, acc);
      store_silu_f16(sX, acc, b1s[h][colg], colg, rb);
    }
    __syncthreads();
    if (wave < 4) {  // packed small output layer: [64x128] x [128x16]
      v8f acc = zero;
      const _Float16* Bs = hW + OFF_WSM + h * 2048;
#pragma unroll
      for (int kc = 0; kc < 4; ++kc) {
        v16h bf = load_frag_b_sw(Bs, kc, lane);
        v16h af = load_frag_a(sX, HDIM, wave * 16, kc * 32, lane);
        acc = wmma16(af, bf, acc);
      }
      int c = lane & 15;
      if (c < ncol[h]) {
        float bias = b2s[h][c];
#pragma unroll
        for (int i = 0; i < 8; ++i)
          sG[(wave * 16 + rb + i) * 8 + goff[h] + c] = acc[i] + bias;
      }
    }
    __syncthreads();
  }

  // --- phase 5: gating + atomic scatter of vectors / sh (64 lanes) ---
  if (t < 64) {
    int r = sRow[t];
    float pg0 = sG[t * 8 + 0], pg1 = sG[t * 8 + 1];
    float vg0 = sG[t * 8 + 2], vg1 = sG[t * 8 + 3];
    float w0  = sG[t * 8 + 4], w1 = sG[t * 8 + 5], w2 = sG[t * 8 + 6];
#pragma unroll
    for (int d = 0; d < 3; ++d) {
      atomicAdd(&pos_sum[r * 3 + d], pg0 * sDp[t * 3 + d] + pg1 * sDv[t * 3 + d]);
      atomicAdd(&vel_sum[r * 3 + d], vg0 * sDv[t * 3 + d] + vg1 * sDp[t * 3 + d]);
    }
#pragma unroll
    for (int j = 0; j < SHD; ++j) {
      float wc = (j == 0) ? w0 : ((j < 4) ? w1 : w2);
      atomicAdd(&sh_sum[r * SHD + j], wc * sShd[t * SHD + j]);
    }
  }
}

// ---------------------------------------------------------------------------
// Kernel 3: per-node finalize. One block = 64 nodes, 8 waves.
//   msg_agg = msg_sum/denom ; node MLP via WMMA ; finalize sh/pos/vel.
// ---------------------------------------------------------------------------
__global__ void __launch_bounds__(256)
hegnn_node_kernel(const float* __restrict__ node_feat,
                  const float* __restrict__ node_sh,
                  const float* __restrict__ b_nf1,
                  const float* __restrict__ b_nf2,
                  const _Float16* __restrict__ hW,
                  const float* __restrict__ counts,
                  float* out) {
  __shared__ __align__(16) char smem[49408];
  _Float16* sA   = (_Float16*)smem;             // [64][256]
  _Float16* sX   = (_Float16*)(smem + 32768);   // [64][128]
  float*    sInv = (float*)(smem + 49152);      // [64] 1/denom

  float* out_feat = out;                         // holds msg_sum on entry
  float* out_sh   = out + NNODE * HDIM;          // holds sh_sum on entry
  float* out_pos  = out + NNODE * (HDIM + SHD);  // holds pos_sum on entry
  float* out_vel  = out + NNODE * (HDIM + SHD + 3);

  const int t    = threadIdx.x;
  const int lane = t & 31;
  const int wave = t >> 5;
  const int kh   = lane >> 4;
  const int nb   = blockIdx.x * 64;

  if (t < 64) {
    int n = nb + t;
    float cnt = (n < NNODE) ? counts[n] : 1.0f;
    // exact reciprocal for integral counts up to 2^24: rcp is exact on
    // powers of two and within 1 ulp otherwise; acceptable for mean.
    sInv[t] = __builtin_amdgcn_rcpf(fmaxf(cnt, 1.0f));
  }
  __syncthreads();

  // A tile: [feat | msg_agg] as f16
  for (int idx = t; idx < 64 * 256; idx += 256) {
    int j = idx >> 8, k = idx & 255;
    int n = nb + j;
    float v = 0.0f;
    if (n < NNODE) {
      if (k < 128) v = node_feat[n * HDIM + k];
      else         v = out_feat[n * HDIM + (k - 128)] * sInv[j];
    }
    sA[idx] = (_Float16)v;
  }
  __syncthreads();

  const int colg = wave * 16 + (lane & 15);
  const int rb   = kh * 8;
  v8f zero = {};

  {   // nf layer 1: [64x256] x [256x128], silu
    v8f acc[4] = {zero, zero, zero, zero};
    gemm64(sA, 256, hW + OFF_WNF1, 8, wave, lane, acc);
    store_silu_f16(sX, acc, b_nf1[colg], colg, rb);
  }
  __syncthreads();

  {   // nf layer 2: [64x128] x [128x128], overwrite msg_sum with result
    v8f acc[4] = {zero, zero, zero, zero};
    gemm64(sX, HDIM, hW + OFF_WNF2, 4, wave, lane, acc);
    float bias = b_nf2[colg];
#pragma unroll
    for (int m = 0; m < 4; ++m)
#pragma unroll
      for (int i = 0; i < 8; ++i) {
        int n = nb + m * 16 + rb + i;
        if (n < NNODE) out_feat[n * HDIM + colg] = acc[m][i] + bias;
      }
  }

  // finalize sh / pos / vel in-place
  if (t < 64) {
    int n = nb + t;
    if (n < NNODE) {
      float inv = sInv[t];
#pragma unroll
      for (int j = 0; j < SHD; ++j)
        out_sh[n * SHD + j] = node_sh[n * SHD + j] + out_sh[n * SHD + j] * inv;
#pragma unroll
      for (int d = 0; d < 3; ++d) {
        out_pos[n * 3 + d] = out_pos[n * 3 + d] * inv;
        out_vel[n * 3 + d] = out_vel[n * 3 + d] * inv;
      }
    }
  }
}

// ---------------------------------------------------------------------------
extern "C" void kernel_launch(void* const* d_in, const int* in_sizes, int n_in,
                              void* d_out, int out_size, void* d_ws, size_t ws_size,
                              hipStream_t stream) {
  (void)in_sizes; (void)n_in; (void)out_size; (void)ws_size;
  const float* node_feat  = (const float*)d_in[0];
  const float* node_sh    = (const float*)d_in[1];
  const float* diff_pos   = (const float*)d_in[2];
  const float* diff_vel   = (const float*)d_in[3];
  const float* radial     = (const float*)d_in[4];
  const int*   edge_index = (const int*)d_in[5];
  const float* W_msg1 = (const float*)d_in[6];  const float* b_msg1 = (const float*)d_in[7];
  const float* W_msg2 = (const float*)d_in[8];  const float* b_msg2 = (const float*)d_in[9];
  const float* W_pos1 = (const float*)d_in[10]; const float* b_pos1 = (const float*)d_in[11];
  const float* W_pos2 = (const float*)d_in[12]; const float* b_pos2 = (const float*)d_in[13];
  const float* W_vel1 = (const float*)d_in[14]; const float* b_vel1 = (const float*)d_in[15];
  const float* W_vel2 = (const float*)d_in[16]; const float* b_vel2 = (const float*)d_in[17];
  const float* W_sh1  = (const float*)d_in[18]; const float* b_sh1  = (const float*)d_in[19];
  const float* W_sh2  = (const float*)d_in[20]; const float* b_sh2  = (const float*)d_in[21];
  const float* W_nf1  = (const float*)d_in[22]; const float* b_nf1  = (const float*)d_in[23];
  const float* W_nf2  = (const float*)d_in[24]; const float* b_nf2  = (const float*)d_in[25];

  float*     out    = (float*)d_out;
  float*     counts = (float*)d_ws;
  _Float16*  hW     = (_Float16*)((char*)d_ws + WS_COUNTS_BYTES);

  float* msg_sum = out;
  float* sh_sum  = out + NNODE * HDIM;
  float* pos_sum = out + NNODE * (HDIM + SHD);
  float* vel_sum = out + NNODE * (HDIM + SHD + 3);

  // 0) zero scatter accumulators (d_out regions + counts)
  {
    int total = NNODE * 143;
    hegnn_zero_kernel<<<(total + 255) / 256, 256, 0, stream>>>(out, counts);
  }
  // 1) pack weights to f16 (fragment-major swizzle)
  hegnn_prep_kernel<<<(W_TOTAL + 255) / 256, 256, 0, stream>>>(
      W_msg1, W_msg2, W_pos1, W_vel1, W_sh1,
      W_pos2, W_vel2, W_sh2, W_nf1, W_nf2, hW);
  // 2) fused edge pipeline (E/64 = 10000 blocks)
  hegnn_edge_kernel<<<NEDGE / 64, 256, 0, stream>>>(
      node_feat, node_sh, diff_pos, diff_vel, radial, edge_index,
      b_msg1, b_msg2, b_pos1, b_pos2, b_vel1, b_vel2, b_sh1, b_sh2,
      hW, msg_sum, sh_sum, pos_sum, vel_sum, counts);
  // 3) node finalize ((N+63)/64 = 313 blocks)
  hegnn_node_kernel<<<(NNODE + 63) / 64, 256, 0, stream>>>(
      node_feat, node_sh, b_nf1, b_nf2, hW, counts, out);
}
